// DecoderBlock_14370960573064
// MI455X (gfx1250) — matrix operand order repaired and checked by reading, not compiled, not run on hardware
//
#include <hip/hip_runtime.h>
#include <hip/hip_bf16.h>

// ---------------------------------------------------------------------------
// Decoder block on MI455X (gfx1250, wave32, WMMA f32_16x16x32_f16)
//   B=512 T=128 D=384 H=6 DH=64 DFF=1536, fp32 in/out, f16 WMMA internally.
//   GEMM B-panels staged into LDS by the Tensor Data Mover (TENSORcnt),
//   double-buffered; falls back to global_load_async_to_lds_b128 (ASYNCcnt).
// ---------------------------------------------------------------------------

typedef _Float16 half8  __attribute__((ext_vector_type(8)));
typedef _Float16 half16 __attribute__((ext_vector_type(16)));
typedef float    float8 __attribute__((ext_vector_type(8)));
typedef unsigned int v4u __attribute__((ext_vector_type(4)));
typedef int v4i __attribute__((ext_vector_type(4)));
typedef int v8i __attribute__((ext_vector_type(8)));

union H16 { half16 v; half8 h[2]; };

constexpr int Bsz = 512, Tsz = 128, Dsz = 384, Hn = 6, DHsz = 64, DFFsz = 1536;
constexpr int Mrows = Bsz * Tsz;  // 65536
constexpr int KC = 64;            // K chunk depth staged in LDS

#if __has_builtin(__builtin_amdgcn_tensor_load_to_lds)
#define HAVE_TDM 1
#else
#define HAVE_TDM 0
#endif
#if __has_include(<hip/amd_detail/amd_gfx1250_TDM.h>)
#define TDM_6ARG 1
#else
#define TDM_6ARG 0
#endif

// ---- WMMA fragment loaders (layouts per CDNA5 ISA §7.12.2) ------

// A: 16x32 f16 tile, row-major source with leading dim ld (elements).
__device__ __forceinline__ half16 load_a_frag(const _Float16* __restrict__ base, int ld) {
  const int lane = threadIdx.x & 31;
  const int row  = lane & 15;
  const int kb   = (lane >> 4) * 8;
  H16 r;
  r.h[0] = *(const half8*)(base + (size_t)row * ld + kb);
  r.h[1] = *(const half8*)(base + (size_t)row * ld + 16 + kb);
  return r.v;
}

// B: 32x16 (KxN) tile whose storage is TRANSPOSED: Bt[n][k], leading dim ld.
__device__ __forceinline__ half16 load_bt_frag(const _Float16* __restrict__ base, int ld) {
  const int lane = threadIdx.x & 31;
  const int n    = lane & 15;
  const int kb   = (lane >> 4) * 8;
  H16 r;
  r.h[0] = *(const half8*)(base + (size_t)n * ld + kb);
  r.h[1] = *(const half8*)(base + (size_t)n * ld + 16 + kb);
  return r.v;
}

__device__ __forceinline__ float8 fzero8() {
  float8 z = {0.f, 0.f, 0.f, 0.f, 0.f, 0.f, 0.f, 0.f};
  return z;
}

// LDS byte offset of a __shared__ object: generic LDS addresses carry the
// workgroup-relative offset in addr[31:0] (aperture map, cdna5_isa 07 §10.2).
__device__ __forceinline__ unsigned lds_off(const void* p) {
  return (unsigned)(uintptr_t)p;
}

// Async copy of one 16-byte segment global -> LDS (ASYNCcnt-tracked).
__device__ __forceinline__ void async_g2l_b128(unsigned lds_byte_off, const void* gptr) {
  asm volatile("global_load_async_to_lds_b128 %0, %1, off"
               :: "v"(lds_byte_off), "v"((unsigned long long)(uintptr_t)gptr)
               : "memory");
}

#if HAVE_TDM
// One TDM descriptor load: tile_h rows x tile_w f16 elements, row stride
// row_stride_elems, packed contiguously at lds_byte_off. D# packing per
// CDNA5 ISA (2D tile; groups 2/3 zero).
__device__ __forceinline__ void tdm_load_tile(unsigned lds_byte_off, const void* gptr,
                                              unsigned tile_w, unsigned tile_h,
                                              unsigned row_stride_elems) {
  const unsigned long long ga = (unsigned long long)(uintptr_t)gptr;
  v4u g0;
  g0[0] = 1u;                                               // count=1 (valid D#)
  g0[1] = lds_byte_off;                                     // lds_addr [63:32]
  g0[2] = (unsigned)(ga & 0xffffffffu);                     // global_addr lo
  g0[3] = (unsigned)((ga >> 32) & 0x01ffffffu) | (2u << 30);// addr hi | type=2
  const unsigned td0 = 0x40000000u, td1 = 0x40000000u;      // huge dims: no OOB
  v8i g1;
  g1[0] = (int)(1u << 16);                                  // data_size=1 (2B)
  g1[1] = (int)((td0 & 0xffffu) << 16);                     // tensor_dim0[15:0]
  g1[2] = (int)((td0 >> 16) | ((td1 & 0xffffu) << 16));     // td0 hi | td1 lo
  g1[3] = (int)((td1 >> 16) | (tile_w << 16));              // td1 hi | tile_dim0
  g1[4] = (int)(tile_h & 0xffffu);                          // tile_dim1 (dim2=0)
  g1[5] = (int)row_stride_elems;                            // dim0_stride[31:0]
  g1[6] = 0;                                                // stride hi bits
  g1[7] = 0;
  v4i z4 = {0, 0, 0, 0};
#if TDM_6ARG
  v8i z8 = {0, 0, 0, 0, 0, 0, 0, 0};
  __builtin_amdgcn_tensor_load_to_lds(g0, g1, z4, z4, z8, 0);
#else
  __builtin_amdgcn_tensor_load_to_lds(g0, g1, z4, z4, 0);
#endif
}
#endif

// Issue the staging of one 64-row x KC-halves B chunk into LDS.
__device__ __forceinline__ void stage_issue(unsigned lds_byte_off,
                                            const _Float16* __restrict__ gsrc, int ldb) {
#if HAVE_TDM
  if ((threadIdx.x >> 5) == 0)   // single TDM op, issued by wave 0 only
    tdm_load_tile(lds_byte_off, gsrc, KC, 64, (unsigned)ldb);
#else
#pragma unroll
  for (int j = 0; j < 2; ++j) {
    const int idx = threadIdx.x + j * 256;
    const int n  = idx >> 3;
    const int ks = (idx & 7) * 8;
    async_g2l_b128(lds_byte_off + (unsigned)(n * KC + ks) * 2,
                   gsrc + (size_t)n * ldb + ks);
  }
#endif
}

// Wait until at most `more` staged chunks remain in flight (this wave's ops).
__device__ __forceinline__ void stage_wait(bool more) {
#if HAVE_TDM
  if ((threadIdx.x >> 5) == 0) {
#if __has_builtin(__builtin_amdgcn_s_wait_tensorcnt)
    if (more) __builtin_amdgcn_s_wait_tensorcnt((short)1);
    else      __builtin_amdgcn_s_wait_tensorcnt((short)0);
#else
    if (more) asm volatile("s_wait_tensorcnt 0x1" ::: "memory");
    else      asm volatile("s_wait_tensorcnt 0x0" ::: "memory");
#endif
  }
#else
  if (more) asm volatile("s_wait_asynccnt 0x2" ::: "memory");
  else      asm volatile("s_wait_asynccnt 0x0" ::: "memory");
#endif
}

// ---- weight convert + transpose: W[K][N] fp32 -> Wt[N][K] f16 --------------
__global__ void convT_kernel(const float* __restrict__ W, _Float16* __restrict__ Wt,
                             int K, int N) {
  int idx = blockIdx.x * 256 + threadIdx.x;
  if (idx >= K * N) return;
  int n = idx / K, k = idx % K;
  Wt[idx] = (_Float16)W[(size_t)k * N + n];
}

// ---- LayerNorm: one wave per row of [rows, 384], f16 output ----------------
__global__ __launch_bounds__(256) void ln_kernel(const float* __restrict__ x,
                                                 const float* __restrict__ g,
                                                 const float* __restrict__ b,
                                                 _Float16* __restrict__ out) {
  const int wave = threadIdx.x >> 5;
  const int lane = threadIdx.x & 31;
  const size_t row = (size_t)blockIdx.x * 8 + wave;
  const float* xr = x + row * Dsz;
  float s = 0.f, s2 = 0.f;
#pragma unroll
  for (int c = lane; c < Dsz; c += 32) { float f = xr[c]; s += f; s2 += f * f; }
#pragma unroll
  for (int off = 16; off > 0; off >>= 1) {
    s  += __shfl_xor(s, off, 32);
    s2 += __shfl_xor(s2, off, 32);
  }
  const float mu = s * (1.f / Dsz);
  const float var = s2 * (1.f / Dsz) - mu * mu;
  const float rs = rsqrtf(var + 1e-5f);
  _Float16* orow = out + row * Dsz;
#pragma unroll
  for (int c = lane; c < Dsz; c += 32)
    orow[c] = (_Float16)((xr[c] - mu) * rs * g[c] + b[c]);
}

// ---- Generic WMMA GEMM: out[M,N] = act(A[M,K] @ Bt[N,K]^T + bias) + resid --
// Block = 8 waves, tile 256 rows x 64 cols. Wave w owns rows [.. + 32w, +32)
// (two 16-row tiles -> 8 WMMAs per shared B-fragment set). The 64-col B panel
// is staged per 64-deep K chunk by the TDM, double-buffered.
template <int OUT_F16, int RELU, int HAS_BIAS, int HAS_RES>
__global__ __launch_bounds__(256) void gemm_f16(const _Float16* __restrict__ A, int lda,
                                                const _Float16* __restrict__ Bt, int ldb,
                                                const float* __restrict__ bias,
                                                const float* __restrict__ resid, int ldr,
                                                void* __restrict__ outp, int ldo, int K) {
  __shared__ _Float16 sB[2][64 * KC];  // 2 x 8 KB double buffer, [n][k] ld=KC

  const int wave = threadIdx.x >> 5;
  const int lane = threadIdx.x & 31;
  const size_t rowA = (size_t)blockIdx.y * 256 + wave * 32;
  const int col0 = blockIdx.x * 64;
  const _Float16* bpanel = Bt + (size_t)col0 * ldb;

  float8 acc[2][4];
#pragma unroll
  for (int r = 0; r < 2; ++r)
#pragma unroll
    for (int j = 0; j < 4; ++j) acc[r][j] = fzero8();

  const int nch = K / KC;
  stage_issue(lds_off(sB[0]), bpanel, ldb);

  for (int ic = 0; ic < nch; ++ic) {
    const int kc = ic * KC;
    if (ic + 1 < nch) {
      stage_issue(lds_off(sB[(ic + 1) & 1]), bpanel + kc + KC, ldb);
      stage_wait(true);    // chunk ic complete, chunk ic+1 may stay in flight
    } else {
      stage_wait(false);
    }
    __syncthreads();

    const _Float16* sb = sB[ic & 1];
#pragma unroll
    for (int kk = 0; kk < KC; kk += 32) {
      half16 a0 = load_a_frag(A + (rowA +  0) * lda + kc + kk, lda);
      half16 a1 = load_a_frag(A + (rowA + 16) * lda + kc + kk, lda);
      half16 b0 = load_bt_frag(sb + 0  * KC + kk, KC);
      half16 b1 = load_bt_frag(sb + 16 * KC + kk, KC);
      half16 b2 = load_bt_frag(sb + 32 * KC + kk, KC);
      half16 b3 = load_bt_frag(sb + 48 * KC + kk, KC);
      acc[0][0] = __builtin_amdgcn_wmma_f32_16x16x32_f16(false, a0, false, b0, (short)0, acc[0][0], false, false);
      acc[0][1] = __builtin_amdgcn_wmma_f32_16x16x32_f16(false, a0, false, b1, (short)0, acc[0][1], false, false);
      acc[0][2] = __builtin_amdgcn_wmma_f32_16x16x32_f16(false, a0, false, b2, (short)0, acc[0][2], false, false);
      acc[0][3] = __builtin_amdgcn_wmma_f32_16x16x32_f16(false, a0, false, b3, (short)0, acc[0][3], false, false);
      acc[1][0] = __builtin_amdgcn_wmma_f32_16x16x32_f16(false, a1, false, b0, (short)0, acc[1][0], false, false);
      acc[1][1] = __builtin_amdgcn_wmma_f32_16x16x32_f16(false, a1, false, b1, (short)0, acc[1][1], false, false);
      acc[1][2] = __builtin_amdgcn_wmma_f32_16x16x32_f16(false, a1, false, b2, (short)0, acc[1][2], false, false);
      acc[1][3] = __builtin_amdgcn_wmma_f32_16x16x32_f16(false, a1, false, b3, (short)0, acc[1][3], false, false);
    }
    __syncthreads();  // protect the buffer we just read before re-staging it
  }

  const int hi = lane >> 4, n = lane & 15;
#pragma unroll
  for (int rt = 0; rt < 2; ++rt)
#pragma unroll
    for (int j = 0; j < 4; ++j) {
      const int col = col0 + j * 16 + n;
      const float bv = HAS_BIAS ? bias[col] : 0.f;
#pragma unroll
      for (int e = 0; e < 8; ++e) {
        const size_t row = rowA + rt * 16 + e + 8 * hi;
        float val = acc[rt][j][e] + bv;
        if (HAS_RES) val += resid[row * ldr + col];
        if (RELU) val = fmaxf(val, 0.f);
        if (OUT_F16) ((_Float16*)outp)[row * ldo + col] = (_Float16)val;
        else         ((float*)outp)[row * ldo + col]    = val;
      }
    }
}

// ---- Causal attention for one (b,h): 128x128 scores+softmax in LDS ---------
__global__ __launch_bounds__(256) void attn_kernel(const _Float16* __restrict__ q,
                                                   const _Float16* __restrict__ k,
                                                   const _Float16* __restrict__ v,
                                                   _Float16* __restrict__ o) {
  __shared__ _Float16 sS[Tsz * Tsz];  // 32 KB scores; later reused as V^T
  __shared__ _Float16 sP[Tsz * Tsz];  // 32 KB probabilities -> 64 KB total

  const int b = blockIdx.x / Hn;
  const int h = blockIdx.x % Hn;
  const size_t base = (size_t)b * Tsz * Dsz + h * DHsz;
  const _Float16* qh = q + base;  // [t][dh], stride Dsz
  const _Float16* kh = k + base;
  const _Float16* vh = v + base;

  const int wave = threadIdx.x >> 5;
  const int lane = threadIdx.x & 31;
  const int t0 = wave * 16;
  const int n  = lane & 15;
  const int hi = lane >> 4;

  // zero P (masked region must contribute 0 in P@V)
  {
    unsigned int* p = (unsigned int*)sP;
    for (int i = threadIdx.x; i < Tsz * Tsz / 2; i += 256) p[i] = 0u;
  }

  // ---- scores = (Q @ K^T) * DH^-1/2, lower-triangular tiles only ----
  half16 aq0 = load_a_frag(qh + (size_t)t0 * Dsz + 0, Dsz);
  half16 aq1 = load_a_frag(qh + (size_t)t0 * Dsz + 32, Dsz);
  for (int c = 0; c <= wave; ++c) {
    float8 acc = fzero8();
    half16 b0 = load_bt_frag(kh + (size_t)(c * 16) * Dsz + 0, Dsz);
    acc = __builtin_amdgcn_wmma_f32_16x16x32_f16(false, aq0, false, b0, (short)0, acc, false, false);
    half16 b1 = load_bt_frag(kh + (size_t)(c * 16) * Dsz + 32, Dsz);
    acc = __builtin_amdgcn_wmma_f32_16x16x32_f16(false, aq1, false, b1, (short)0, acc, false, false);
#pragma unroll
    for (int e = 0; e < 8; ++e)
      sS[(t0 + e + 8 * hi) * Tsz + c * 16 + n] = (_Float16)(acc[e] * 0.125f);
  }
  __syncthreads();

  // ---- causal softmax, one thread per row ----
  if (threadIdx.x < Tsz) {
    const int t = threadIdx.x;
    const _Float16* srow = sS + t * Tsz;
    _Float16* prow = sP + t * Tsz;
    float m = -1e30f;
    for (int s = 0; s <= t; ++s) m = fmaxf(m, (float)srow[s]);
    float sum = 0.f;
    for (int s = 0; s <= t; ++s) sum += __expf((float)srow[s] - m);
    const float inv = 1.f / sum;
    for (int s = 0; s <= t; ++s) prow[s] = (_Float16)(__expf((float)srow[s] - m) * inv);
  }
  __syncthreads();

  // ---- stage V^T into the dead score buffer: sVt[d][s], ld = Tsz (16 KB) ----
  _Float16* sVt = sS;
  for (int i = threadIdx.x; i < DHsz * Tsz; i += 256) {
    const int s = i >> 6;   // 0..127  (i = s*64 + d; global read coalesced)
    const int d = i & 63;
    sVt[d * Tsz + s] = vh[(size_t)s * Dsz + d];
  }
  __syncthreads();

  // ---- O = P @ V  (A from sP, B from sVt: both pure LDS traffic) ----
  float8 acc0 = fzero8(), acc1 = fzero8(), acc2 = fzero8(), acc3 = fzero8();
  for (int s0 = 0; s0 <= t0; s0 += 32) {  // P is zero beyond the diagonal
    half16 a  = load_a_frag(sP + (size_t)t0 * Tsz + s0, Tsz);
    half16 b0 = load_bt_frag(sVt + 0  * Tsz + s0, Tsz);
    half16 b1 = load_bt_frag(sVt + 16 * Tsz + s0, Tsz);
    half16 b2 = load_bt_frag(sVt + 32 * Tsz + s0, Tsz);
    half16 b3 = load_bt_frag(sVt + 48 * Tsz + s0, Tsz);
    acc0 = __builtin_amdgcn_wmma_f32_16x16x32_f16(false, a, false, b0, (short)0, acc0, false, false);
    acc1 = __builtin_amdgcn_wmma_f32_16x16x32_f16(false, a, false, b1, (short)0, acc1, false, false);
    acc2 = __builtin_amdgcn_wmma_f32_16x16x32_f16(false, a, false, b2, (short)0, acc2, false, false);
    acc3 = __builtin_amdgcn_wmma_f32_16x16x32_f16(false, a, false, b3, (short)0, acc3, false, false);
  }
  float8 accs[4] = {acc0, acc1, acc2, acc3};
  _Float16* oh = o + base;
#pragma unroll
  for (int j = 0; j < 4; ++j)
#pragma unroll
    for (int e = 0; e < 8; ++e)
      oh[(size_t)(t0 + e + 8 * hi) * Dsz + j * 16 + n] = (_Float16)accs[j][e];
}

// ---------------------------------------------------------------------------
extern "C" void kernel_launch(void* const* d_in, const int* in_sizes, int n_in,
                              void* d_out, int out_size, void* d_ws, size_t ws_size,
                              hipStream_t stream) {
  const float* x   = (const float*)d_in[0];
  const float* Wq  = (const float*)d_in[1];
  const float* Wk  = (const float*)d_in[2];
  const float* Wv  = (const float*)d_in[3];
  const float* Wo  = (const float*)d_in[4];
  const float* bo  = (const float*)d_in[5];
  const float* W1  = (const float*)d_in[6];
  const float* b1  = (const float*)d_in[7];
  const float* W2  = (const float*)d_in[8];
  const float* b2  = (const float*)d_in[9];
  const float* g1  = (const float*)d_in[10];
  const float* be1 = (const float*)d_in[11];
  const float* g2  = (const float*)d_in[12];
  const float* be2 = (const float*)d_in[13];

  char* ws = (char*)d_ws;
  size_t off = 0;
  auto alloc = [&](size_t bytes) -> void* {
    void* p = ws + off;
    off += (bytes + 255) & ~(size_t)255;
    return p;
  };

  const size_t M = Mrows;
  _Float16* hbuf = (_Float16*)alloc(M * Dsz * 2);   // LN1 out, later LN2 out
  _Float16* qb   = (_Float16*)alloc(M * Dsz * 2);
  _Float16* kb   = (_Float16*)alloc(M * Dsz * 2);
  _Float16* vb   = (_Float16*)alloc(M * Dsz * 2);
  _Float16* ob   = (_Float16*)alloc(M * Dsz * 2);
  float*    x1   = (float*)alloc(M * Dsz * 4);      // attn residual output
  _Float16* WqT  = (_Float16*)alloc((size_t)Dsz * Dsz * 2);
  _Float16* WkT  = (_Float16*)alloc((size_t)Dsz * Dsz * 2);
  _Float16* WvT  = (_Float16*)alloc((size_t)Dsz * Dsz * 2);
  _Float16* WoT  = (_Float16*)alloc((size_t)Dsz * Dsz * 2);
  _Float16* W1T  = (_Float16*)alloc((size_t)Dsz * DFFsz * 2);
  _Float16* W2T  = (_Float16*)alloc((size_t)DFFsz * Dsz * 2);
  // ff1 (M x DFF f16 = 201.3 MB) aliases the dead q/k/v/o region (4 x 50.33 MB)
  _Float16* ff1 = qb;

  // 1) weight transpose/convert to f16 [N][K]
  convT_kernel<<<(Dsz * Dsz + 255) / 256, 256, 0, stream>>>(Wq, WqT, Dsz, Dsz);
  convT_kernel<<<(Dsz * Dsz + 255) / 256, 256, 0, stream>>>(Wk, WkT, Dsz, Dsz);
  convT_kernel<<<(Dsz * Dsz + 255) / 256, 256, 0, stream>>>(Wv, WvT, Dsz, Dsz);
  convT_kernel<<<(Dsz * Dsz + 255) / 256, 256, 0, stream>>>(Wo, WoT, Dsz, Dsz);
  convT_kernel<<<(Dsz * DFFsz + 255) / 256, 256, 0, stream>>>(W1, W1T, Dsz, DFFsz);
  convT_kernel<<<(DFFsz * Dsz + 255) / 256, 256, 0, stream>>>(W2, W2T, DFFsz, Dsz);

  // 2) LN1
  ln_kernel<<<M / 8, 256, 0, stream>>>(x, g1, be1, hbuf);

  // 3) Q,K,V projections (f16 out)
  dim3 g384(Dsz / 64, M / 256);
  gemm_f16<1, 0, 0, 0><<<g384, 256, 0, stream>>>(hbuf, Dsz, WqT, Dsz, nullptr, nullptr, 0, qb, Dsz, Dsz);
  gemm_f16<1, 0, 0, 0><<<g384, 256, 0, stream>>>(hbuf, Dsz, WkT, Dsz, nullptr, nullptr, 0, kb, Dsz, Dsz);
  gemm_f16<1, 0, 0, 0><<<g384, 256, 0, stream>>>(hbuf, Dsz, WvT, Dsz, nullptr, nullptr, 0, vb, Dsz, Dsz);

  // 4) causal attention per (b,h)
  attn_kernel<<<Bsz * Hn, 256, 0, stream>>>(qb, kb, vb, ob);

  // 5) x1 = x + O @ Wo + bo  (fp32 out)
  gemm_f16<0, 0, 1, 1><<<g384, 256, 0, stream>>>(ob, Dsz, WoT, Dsz, bo, x, Dsz, x1, Dsz, Dsz);

  // 6) LN2
  ln_kernel<<<M / 8, 256, 0, stream>>>(x1, g2, be2, hbuf);

  // 7) ff1 = relu(h2 @ W1 + b1)  (f16 out)
  dim3 g1536(DFFsz / 64, M / 256);
  gemm_f16<1, 1, 1, 0><<<g1536, 256, 0, stream>>>(hbuf, Dsz, W1T, Dsz, b1, nullptr, 0, ff1, DFFsz, Dsz);

  // 8) out = x1 + ff1 @ W2 + b2  (fp32 out)
  gemm_f16<0, 0, 1, 1><<<g384, 256, 0, stream>>>(ff1, DFFsz, W2T, DFFsz, b2, x1, Dsz, (float*)d_out, Dsz, DFFsz);

  (void)in_sizes; (void)n_in; (void)out_size; (void)ws_size;
}
